// Decoder_70677981823491
// MI455X (gfx1250) — matrix operand order repaired and checked by reading, not compiled, not run on hardware
//
#include <hip/hip_runtime.h>
#include <hip/hip_bf16.h>
#include <math.h>
#include <stdint.h>

// ---------------------------------------------------------------------------
// CDNA5 (gfx1250) WMMA types
// ---------------------------------------------------------------------------
typedef __attribute__((ext_vector_type(16))) __bf16 v16bf;
typedef __attribute__((ext_vector_type(8)))  __bf16 v8bf;
typedef __attribute__((ext_vector_type(8)))  float  v8f;
typedef unsigned int u32x4 __attribute__((ext_vector_type(4)));
typedef int          i32x4 __attribute__((ext_vector_type(4)));
typedef int          i32x8 __attribute__((ext_vector_type(8)));

#define DIMM   768
#define INNERD 768
#define FFD    3072
#define BB     2
#define NN     2048
#define KCH    32
#define HEADS  12
#define DHEAD  64
#define PAD63  63
#define SEQC   64           // chunk query length
#define LKX    256          // R*CN cross keys per chunk
#define GCROSS (BB*KCH)     // 64 chunks
#define ROWS   (BB*NN)      // 4096
#define CROWS  (GCROSS*LKX) // 16384

#if __has_builtin(__builtin_amdgcn_tensor_load_to_lds) && __has_builtin(__builtin_amdgcn_s_wait_tensorcnt)
#define HAVE_TDM 1
#else
#define HAVE_TDM 0
#endif

__device__ __forceinline__ __bf16 f2bf(float x) { return (__bf16)x; }

__device__ __forceinline__ v8f wmma_bf16(v16bf a, v16bf b, v8f c) {
  // emits v_wmma_f32_16x16x32_bf16
  return __builtin_amdgcn_wmma_f32_16x16x32_bf16(false, a, false, b, (short)0, c,
                                                 false, false);
}

// A-fragment (16x32 bf16): lane<16 holds row=lane, K in {0..15};
// lane>=16 holds K in {16..31} (kb0 = 16). Two 16B loads.
__device__ __forceinline__ v16bf frag_a(const __bf16* row, int kb0) {
  v8bf lo = *(const v8bf*)(row + kb0);
  v8bf hi = *(const v8bf*)(row + kb0 + 8);
  v16bf r;
#pragma unroll
  for (int i = 0; i < 8; ++i) { r[i] = lo[i]; r[i + 8] = hi[i]; }
  return r;
}

// B-fragment column: 16 contiguous K values starting at kbl (0 / 16 per half-wave).
__device__ __forceinline__ v16bf frag_contig(const __bf16* p) {
  v8bf lo = *(const v8bf*)p;
  v8bf hi = *(const v8bf*)(p + 8);
  v16bf r;
#pragma unroll
  for (int i = 0; i < 8; ++i) { r[i] = lo[i]; r[i + 8] = hi[i]; }
  return r;
}

// ---------------------------------------------------------------------------
// Elementwise fp32 -> bf16 cast (retrieved context, converted once per call)
// ---------------------------------------------------------------------------
__global__ void k_cast_bf16(const float* __restrict__ in, __bf16* __restrict__ out, int n) {
  int i = blockIdx.x * blockDim.x + threadIdx.x;
  if (i < n) out[i] = f2bf(in[i]);
}

// ---------------------------------------------------------------------------
// Weight transpose+cast: fp32 W[K][N] -> bf16 WT[N][K]. 32x32 LDS tile.
// ---------------------------------------------------------------------------
__global__ void __launch_bounds__(256)
k_transpose_bf16(const float* __restrict__ in, __bf16* __restrict__ out, int K, int N) {
  __shared__ float t[32][33];
  int kb = blockIdx.x * 32, nb = blockIdx.y * 32;
  int tx = threadIdx.x & 31, ty = threadIdx.x >> 5;  // 32 x 8
#pragma unroll
  for (int i = 0; i < 4; ++i) {
    int k = kb + ty + i * 8;
    t[ty + i * 8][tx] = in[(size_t)k * N + nb + tx];
  }
  __syncthreads();
#pragma unroll
  for (int i = 0; i < 4; ++i) {
    int n = nb + ty + i * 8;
    out[(size_t)n * K + kb + tx] = f2bf(t[tx][ty + i * 8]);
  }
}

// ---------------------------------------------------------------------------
// RMSNorm: out = x / max(||x||/sqrt(D), eps) * gamma (one block per row).
// ---------------------------------------------------------------------------
__global__ void k_rmsnorm(const float* __restrict__ x, const float* __restrict__ gamma,
                          float* __restrict__ outf, __bf16* __restrict__ outb) {
  int row = blockIdx.x;
  const float* xr = x + (size_t)row * DIMM;
  float ss = 0.f;
  for (int c = threadIdx.x; c < DIMM; c += blockDim.x) { float v = xr[c]; ss += v * v; }
  __shared__ float red[256];
  red[threadIdx.x] = ss;
  __syncthreads();
  for (int s = 128; s > 0; s >>= 1) {
    if ((int)threadIdx.x < s) red[threadIdx.x] += red[threadIdx.x + s];
    __syncthreads();
  }
  float n   = sqrtf(red[0]) * 0.03608439182435161f; // 1/sqrt(768)
  float inv = 1.f / fmaxf(n, 1e-8f);
  for (int c = threadIdx.x; c < DIMM; c += blockDim.x) {
    float v = xr[c] * inv * gamma[c];
    if (outf) outf[(size_t)row * DIMM + c] = v;
    if (outb) outb[(size_t)row * DIMM + c] = f2bf(v);
  }
}

// ---------------------------------------------------------------------------
// WMMA GEMM: C[M,N] = A[M,K] @ W[K,N], both operands bf16, fp32 accumulate.
// A: bf16 row-major, direct per-wave fragment loads (2 x b128 / K-step).
// BT: pre-transposed bf16 weight [N][K]; 64x32 tile pulled into LDS by the
// Tensor Data Mover (wave 0 issues TENSOR_LOAD_TO_LDS with a D# describing a
// 32-elem-wide, 64-row tile with 16B LDS row padding -> matches LDK=40),
// tracked by TENSORcnt. Fallback: plain b128 copy loop.
// ---------------------------------------------------------------------------
#define GBM 64
#define GBN 64
#define GBK 32
#define LDK 40  // padded row (80B, keeps 16B alignment, spreads LDS banks)

#if HAVE_TDM
__device__ __forceinline__ void tdm_load_b_tile(const __bf16* gsrc, unsigned lds_base,
                                                int K, int N_) {
  unsigned long long ga = (unsigned long long)(uintptr_t)gsrc;
  u32x4 g0;
  g0[0] = 1u;                                   // count=1 user descriptor
  g0[1] = lds_base;                             // lds_addr (bytes)
  g0[2] = (unsigned)(ga & 0xFFFFFFFFu);         // global_addr[31:0]
  g0[3] = (unsigned)((ga >> 32) & 0x1FFFFFFu)   // global_addr[56:32]
          | (2u << 30);                         // type = 2 ("image")
  i32x8 g1;
  g1[0] = (1 << 16)                             // data_size = 2 bytes
        | (1 << 20)                             // pad_enable
        | (3 << 22)                             // pad_interval: 16 DWORDs (64B row)
        | (3 << 25);                            // pad_amount: 4 DWORDs (16B)
  g1[1] = (K & 0xFFFF) << 16;                   // tensor_dim0[15:0] @ bit48
  g1[2] = ((K >> 16) & 0xFFFF)                  // tensor_dim0[31:16]
        | ((N_ & 0xFFFF) << 16);                // tensor_dim1[15:0]
  g1[3] = ((N_ >> 16) & 0xFFFF)                 // tensor_dim1[31:16]
        | (GBK << 16);                          // tile_dim0 = 32 (contig K)
  g1[4] = GBN;                                  // tile_dim1 = 64 rows, tile_dim2 = 0
  g1[5] = K;                                    // tensor_dim0_stride[31:0] (elems)
  g1[6] = 0;                                    // stride hi / dim1_stride
  g1[7] = 0;
  i32x4 z4 = {0, 0, 0, 0};
#if __clang_major__ >= 23
  i32x8 z8 = {0, 0, 0, 0, 0, 0, 0, 0};
  __builtin_amdgcn_tensor_load_to_lds(g0, g1, z4, z4, z8, 0);
#else
  __builtin_amdgcn_tensor_load_to_lds(g0, g1, z4, z4, 0);
#endif
}
#endif

__global__ void __launch_bounds__(128)
k_gemm(const __bf16* __restrict__ A, const __bf16* __restrict__ BT,
       const float* __restrict__ bias, const float* __restrict__ residual,
       float* __restrict__ Cf, __bf16* __restrict__ Cbf,
       int M, int N, int K, int gelu) {
  (void)M;
  __shared__ __bf16 sB[GBN][LDK];  // sB[n][k]
  int m0 = blockIdx.x * GBM;
  int n0 = blockIdx.y * GBN;
  int tid = threadIdx.x;
  int w = tid >> 5, lane = tid & 31;
  int l16 = lane & 15;
  int kb0 = (lane < 16) ? 0 : 16;
  int kbl = (lane < 16) ? 0 : 16;

  const __bf16* arow = A + (size_t)(m0 + w * 16 + l16) * K;
#if HAVE_TDM
  unsigned lds_base = (unsigned)(uintptr_t)&sB[0][0];  // flat LDS addr[31:0] = LDS offset
#endif

  v8f acc[4];
#pragma unroll
  for (int t = 0; t < 4; ++t) acc[t] = (v8f){0.f,0.f,0.f,0.f,0.f,0.f,0.f,0.f};

  for (int k0 = 0; k0 < K; k0 += GBK) {
#if HAVE_TDM
    if (w == 0) {  // wave 0 drives the Tensor Data Mover
      tdm_load_b_tile(&BT[(size_t)n0 * K + k0], lds_base, K, N);
      __builtin_amdgcn_s_wait_tensorcnt(0);
    }
#else
#pragma unroll
    for (int i = 0; i < 2; ++i) {
      int c = tid + i * 128;
      int n = c >> 2, kc = (c & 3) * 8;
      const __bf16* src = &BT[(size_t)(n0 + n) * K + k0 + kc];
      *(v8bf*)&sB[n][kc] = *(const v8bf*)src;
    }
#endif
    __builtin_prefetch(arow + k0 + GBK, 0, 1);  // next A chunk -> global_prefetch_b8
    __syncthreads();
    v16bf a = frag_a(arow + k0, kb0);
#pragma unroll
    for (int t = 0; t < 4; ++t) {
      v16bf b = frag_contig(&sB[t * 16 + l16][kbl]);
      acc[t] = wmma_bf16(a, b, acc[t]);
    }
    __syncthreads();
  }

  int rowoff = (lane < 16) ? 0 : 8;
#pragma unroll
  for (int t = 0; t < 4; ++t) {
#pragma unroll
    for (int r = 0; r < 8; ++r) {
      int grow = m0 + w * 16 + rowoff + r;   // C layout: vgpr r -> row, lane -> col
      int gcol = n0 + t * 16 + l16;
      float v = acc[t][r];
      if (bias)     v += bias[gcol];
      if (residual) v += residual[(size_t)grow * N + gcol];
      if (gelu)     v = 0.5f * v * (1.f + erff(v * 0.7071067811865475f));
      if (Cbf) Cbf[(size_t)grow * N + gcol] = f2bf(v);
      else     Cf [(size_t)grow * N + gcol] = v;
    }
  }
}

// ---------------------------------------------------------------------------
// RoPE + head split (bf16 -> bf16): src[rows, stride]+off -> heads.
// tdst=0: dst[(g*H+h), L, 64]   (Q / K layout)
// tdst=1: dst[(g*H+h), 64, L]   (V transposed so flash B-frags are contiguous)
// mode 0: pos = local; mode 1: pos = local==0 ? 126 : 0; mode 2: pos = local%128.
// ---------------------------------------------------------------------------
__global__ void k_rope_heads(const __bf16* __restrict__ src, int stride, int off,
                             __bf16* __restrict__ dst, int rows, int L,
                             int mode, float scale, int do_rope, int tdst) {
  int idx = blockIdx.x * blockDim.x + threadIdx.x;
  if (idx >= rows * INNERD) return;
  int row = idx / INNERD;
  int c = idx - row * INNERD;
  int h = c >> 6, d = c & 63;
  size_t sbase = (size_t)row * stride + off;
  float val = (float)src[sbase + c] * scale;
  int local = row % L;
  if (do_rope && d < 32) {
    int d2 = (d < 16) ? d + 16 : d - 16;
    float part = (float)src[sbase + h * 64 + d2] * scale;
    float pos;
    if (mode == 0)      pos = (float)local;
    else if (mode == 1) pos = (local == 0) ? 126.f : 0.f;
    else                pos = (float)(local & 127);
    float ang = pos * __powf(10000.f, -(float)(d & 15) * 0.0625f);
    float rot = (d < 16) ? -part : part;
    val = val * __cosf(ang) + rot * __sinf(ang);
  }
  size_t gh = (size_t)(row / L) * HEADS + h;
  if (tdst) dst[(gh * DHEAD + d) * L + local] = f2bf(val);
  else      dst[(gh * L + local) * DHEAD + d] = f2bf(val);
}

// ---------------------------------------------------------------------------
// Flash attention: one wave per 16-query tile, 32-key blocks, all-WMMA.
// qh/kh: [(g*H+h), L, 64]; vh: [(g*H+h), 64, Lk] (transposed).
// out: bf16 [g*Lq + row, INNER].
// ---------------------------------------------------------------------------
__global__ void __launch_bounds__(32)
k_flash(const __bf16* __restrict__ qh, const __bf16* __restrict__ kh,
        const __bf16* __restrict__ vh, __bf16* __restrict__ out,
        const unsigned char* __restrict__ mask, int Lq, int Lk, int causal) {
  int qtiles = Lq >> 4;
  int qt = blockIdx.x % qtiles;
  int gh = blockIdx.x / qtiles;
  int g = gh / HEADS, h = gh % HEADS;
  int lane = threadIdx.x;
  int l16 = lane & 15;
  int kb0 = (lane < 16) ? 0 : 16;
  int kbl = (lane < 16) ? 0 : 16;
  int rowoff = (lane < 16) ? 0 : 8;

  const __bf16* qbase  = qh + ((size_t)gh * Lq + qt * 16) * DHEAD;
  const __bf16* kbase  = kh + (size_t)gh * Lk * DHEAD;
  const __bf16* vheadT = vh + (size_t)gh * DHEAD * Lk;
  const unsigned char* mrow = mask ? (mask + (size_t)g * Lk) : nullptr;

  v16bf a_lo = frag_a(qbase + l16 * DHEAD, kb0);       // features 0..31
  v16bf a_hi = frag_a(qbase + l16 * DHEAD + 32, kb0);  // features 32..63

  v8f o[4];
#pragma unroll
  for (int t = 0; t < 4; ++t) o[t] = (v8f){0.f,0.f,0.f,0.f,0.f,0.f,0.f,0.f};
  float mr[8], lr[8];
#pragma unroll
  for (int r = 0; r < 8; ++r) { mr[r] = -3.0e38f; lr[r] = 0.f; }

  __shared__ __bf16 pl[16 * 32];  // P tile transpose buffer

  int nkb = Lk >> 5;
  if (causal) {
    int kmax = ((qt * 16 + 15) >> 5) + 1;
    if (kmax < nkb) nkb = kmax;
  }

  for (int kb = 0; kb < nkb; ++kb) {
    const __bf16* kblk = kbase + (size_t)(kb * 32) * DHEAD;
    v8f s[2];
    s[0] = (v8f){0.f,0.f,0.f,0.f,0.f,0.f,0.f,0.f};
    s[1] = (v8f){0.f,0.f,0.f,0.f,0.f,0.f,0.f,0.f};
#pragma unroll
    for (int t = 0; t < 2; ++t) {            // S = Q @ K^T (two 16-key subtiles)
      const __bf16* krow = kblk + (t * 16 + l16) * DHEAD;
      s[t] = wmma_bf16(a_lo, frag_contig(krow + kbl), s[t]);
      s[t] = wmma_bf16(a_hi, frag_contig(krow + 32 + kbl), s[t]);
    }
#pragma unroll
    for (int t = 0; t < 2; ++t) {            // causal / context masks
      int key = kb * 32 + t * 16 + l16;
      bool keep = mrow ? (mrow[key] != 0) : true;
#pragma unroll
      for (int r = 0; r < 8; ++r) {
        int qrow = qt * 16 + rowoff + r;
        if (!keep || (causal && key > qrow)) s[t][r] = -3.0e38f;
      }
    }
#pragma unroll
    for (int r = 0; r < 8; ++r) {            // online softmax (vgpr slot = row)
      float v0 = s[0][r], v1 = s[1][r];
      float mx = fmaxf(v0, v1);
      for (int offi = 1; offi < 16; offi <<= 1) mx = fmaxf(mx, __shfl_xor(mx, offi, 32));
      float mnew = fmaxf(mr[r], mx);
      float scl = __expf(fminf(mr[r] - mnew, 0.f));
      float p0 = __expf(v0 - mnew);
      float p1 = __expf(v1 - mnew);
      float sum = p0 + p1;
      for (int offi = 1; offi < 16; offi <<= 1) sum += __shfl_xor(sum, offi, 32);
      lr[r] = lr[r] * scl + sum;
      mr[r] = mnew;
#pragma unroll
      for (int t = 0; t < 4; ++t) o[t][r] *= scl;
      pl[(rowoff + r) * 32 + l16]      = f2bf(p0);  // transpose P via LDS
      pl[(rowoff + r) * 32 + 16 + l16] = f2bf(p1);
    }
    __syncthreads();
    v16bf ap = frag_a(&pl[l16 * 32], kb0);
#pragma unroll
    for (int t = 0; t < 4; ++t) {            // O += P @ V, V pre-transposed
      const __bf16* vcol = vheadT + (size_t)(t * 16 + l16) * Lk + kb * 32 + kbl;
      o[t] = wmma_bf16(ap, frag_contig(vcol), o[t]);
    }
    __syncthreads();
  }

#pragma unroll
  for (int r = 0; r < 8; ++r) {
    float inv = 1.f / fmaxf(lr[r], 1e-30f);
    int qrow = qt * 16 + rowoff + r;
    __bf16* orow = out + ((size_t)g * Lq + qrow) * INNERD + h * DHEAD;
#pragma unroll
    for (int t = 0; t < 4; ++t) orow[t * 16 + l16] = f2bf(o[t][r] * inv);
  }
}

// ---------------------------------------------------------------------------
// Chunk shifts for cross attention (fp32 residual stream)
// ---------------------------------------------------------------------------
__global__ void k_shift_left(const float* __restrict__ x, float* __restrict__ xs) {
  int idx = blockIdx.x * blockDim.x + threadIdx.x;
  if (idx >= ROWS * DIMM) return;
  int row = idx / DIMM, c = idx - row * DIMM;
  int b = row / NN, n = row - b * NN;
  xs[idx] = (n + PAD63 < NN) ? x[((size_t)b * NN + n + PAD63) * DIMM + c] : 0.f;
}

__global__ void k_shift_right_add(const float* __restrict__ co, float* __restrict__ x) {
  int idx = blockIdx.x * blockDim.x + threadIdx.x;
  if (idx >= ROWS * DIMM) return;
  int row = idx / DIMM, c = idx - row * DIMM;
  int b = row / NN, n = row - b * NN;
  if (n >= PAD63) x[idx] += co[((size_t)b * NN + n - PAD63) * DIMM + c];
}

// ---------------------------------------------------------------------------
// Host orchestration
// ---------------------------------------------------------------------------
extern "C" void kernel_launch(void* const* d_in, const int* in_sizes, int n_in,
                              void* d_out, int out_size, void* d_ws, size_t ws_size,
                              hipStream_t stream) {
  (void)in_sizes; (void)n_in; (void)out_size; (void)ws_size;
  const float* x_in      = (const float*)d_in[0];
  const float* retrieved = (const float*)d_in[1];
  const unsigned char* cmask = (const unsigned char*)d_in[2];
  const float* attn_gamma = (const float*)d_in[3];
  const float* attn_wq    = (const float*)d_in[4];
  const float* attn_wkv   = (const float*)d_in[5];
  const float* attn_wo    = (const float*)d_in[6];
  const float* attn_bo    = (const float*)d_in[7];
  const float* ca_gamma   = (const float*)d_in[8];
  const float* ca_wq      = (const float*)d_in[9];
  const float* ca_wkv     = (const float*)d_in[10];
  const float* ca_wo      = (const float*)d_in[11];
  const float* ca_bo      = (const float*)d_in[12];
  const float* ff_gamma   = (const float*)d_in[13];
  const float* ff_w1      = (const float*)d_in[14];
  const float* ff_b1      = (const float*)d_in[15];
  const float* ff_w2      = (const float*)d_in[16];
  const float* ff_b2      = (const float*)d_in[17];
  const float* final_gamma= (const float*)d_in[18];

  char* wsb = (char*)d_ws;
  size_t off = 0;
  auto alloc = [&](size_t bytes) { size_t o = off; off += (bytes + 255) & ~(size_t)255; return o; };
  float*  X   = (float*)(wsb + alloc((size_t)ROWS * DIMM * 4));        // residual (fp32)
  float*  XNf = (float*)(wsb + alloc((size_t)ROWS * DIMM * 4));        // norm out / co
  float*  XSf = (float*)(wsb + alloc((size_t)ROWS * DIMM * 4));        // shifted x / q-proj
  __bf16* AO  = (__bf16*)(wsb + alloc((size_t)ROWS * DIMM * 2));       // attn out / cross-q
  __bf16* TB  = (__bf16*)(wsb + alloc((size_t)CROWS * 2 * INNERD * 2));// kv / ff hidden
  __bf16* RBF = (__bf16*)(wsb + alloc((size_t)CROWS * DIMM * 2));      // retrieved bf16
  __bf16* WT  = (__bf16*)(wsb + alloc((size_t)DIMM * FFD * 2));        // transposed weight
  __bf16* QH  = (__bf16*)(wsb + alloc((size_t)ROWS * INNERD * 2));
  __bf16* KH  = (__bf16*)(wsb + alloc((size_t)CROWS * INNERD * 2));
  __bf16* VH  = (__bf16*)(wsb + alloc((size_t)CROWS * INNERD * 2));
  __bf16* XNb = (__bf16*)XNf;
  __bf16* XSb = (__bf16*)XSf;

  hipMemcpyAsync(X, x_in, (size_t)ROWS * DIMM * 4, hipMemcpyDeviceToDevice, stream);
  k_cast_bf16<<<(CROWS * DIMM + 255) / 256, 256, 0, stream>>>(retrieved, RBF, CROWS * DIMM);

  const int EWB = (ROWS * DIMM + 255) / 256;
  auto wtrans = [&](const float* w, int K, int N) {
    dim3 g(K / 32, N / 32);
    k_transpose_bf16<<<g, 256, 0, stream>>>(w, WT, K, N);
  };
  auto gemm = [&](const __bf16* A, const float* bias, const float* res,
                  float* Cf, __bf16* Cb, int M, int N, int K, int gelu) {
    dim3 g(M / 64, N / 64), b(128);
    k_gemm<<<g, b, 0, stream>>>(A, WT, bias, res, Cf, Cb, M, N, K, gelu);
  };
  auto rope = [&](const __bf16* src, int stride, int soff, __bf16* dst,
                  int rows, int L, int mode, float scale, int dorope, int tdst) {
    int total = rows * INNERD;
    k_rope_heads<<<(total + 255) / 256, 256, 0, stream>>>(src, stride, soff, dst,
                                                          rows, L, mode, scale, dorope, tdst);
  };

  for (int l = 0; l < 4; ++l) {
    const float* wq  = attn_wq  + (size_t)l * DIMM * INNERD;
    const float* wkv = attn_wkv + (size_t)l * DIMM * 2 * INNERD;
    const float* wo  = attn_wo  + (size_t)l * INNERD * DIMM;
    const float* bo  = attn_bo  + (size_t)l * DIMM;
    const float* cwq = ca_wq    + (size_t)l * DIMM * INNERD;
    const float* cwkv= ca_wkv   + (size_t)l * DIMM * 2 * INNERD;
    const float* cwo = ca_wo    + (size_t)l * INNERD * DIMM;
    const float* cbo = ca_bo    + (size_t)l * DIMM;
    const float* w1  = ff_w1    + (size_t)l * DIMM * FFD;
    const float* b1  = ff_b1    + (size_t)l * FFD;
    const float* w2  = ff_w2    + (size_t)l * FFD * DIMM;
    const float* b2  = ff_b2    + (size_t)l * DIMM;

    // ---- causal self attention ----
    k_rmsnorm<<<ROWS, 256, 0, stream>>>(X, attn_gamma + (size_t)l * DIMM, nullptr, XNb);
    wtrans(wq, DIMM, INNERD);
    gemm(XNb, nullptr, nullptr, nullptr, XSb, ROWS, INNERD, DIMM, 0);
    wtrans(wkv, DIMM, 2 * INNERD);
    gemm(XNb, nullptr, nullptr, nullptr, TB, ROWS, 2 * INNERD, DIMM, 0);
    rope(XSb, INNERD,     0,      QH, ROWS, NN, 0, 0.125f, 1, 0);
    rope(TB,  2 * INNERD, 0,      KH, ROWS, NN, 0, 1.f,    1, 0);
    rope(TB,  2 * INNERD, INNERD, VH, ROWS, NN, 0, 1.f,    0, 1);  // V transposed
    k_flash<<<BB * HEADS * (NN / 16), 32, 0, stream>>>(QH, KH, VH, AO, nullptr, NN, NN, 1);
    wtrans(wo, INNERD, DIMM);
    gemm(AO, bo, X, X, nullptr, ROWS, DIMM, INNERD, 0);            // + residual

    // ---- chunked cross attention ----
    k_shift_left<<<EWB, 256, 0, stream>>>(X, XSf);
    k_rmsnorm<<<ROWS, 256, 0, stream>>>(XSf, ca_gamma + (size_t)l * DIMM, nullptr, XNb);
    wtrans(cwq, DIMM, INNERD);
    gemm(XNb, nullptr, nullptr, nullptr, AO, ROWS, INNERD, DIMM, 0);
    wtrans(cwkv, DIMM, 2 * INNERD);
    gemm(RBF, nullptr, nullptr, nullptr, TB, CROWS, 2 * INNERD, DIMM, 0);
    rope(AO, INNERD,     0,      QH, ROWS,  SEQC, 1, 0.125f, 1, 0);
    rope(TB, 2 * INNERD, 0,      KH, CROWS, LKX,  2, 1.f,    1, 0);
    rope(TB, 2 * INNERD, INNERD, VH, CROWS, LKX,  2, 1.f,    0, 1);
    k_flash<<<GCROSS * HEADS * (SEQC / 16), 32, 0, stream>>>(QH, KH, VH, AO, cmask,
                                                             SEQC, LKX, 0);
    wtrans(cwo, INNERD, DIMM);
    gemm(AO, cbo, nullptr, XNf, nullptr, ROWS, DIMM, INNERD, 0);   // co -> XNf
    k_shift_right_add<<<EWB, 256, 0, stream>>>(XNf, X);

    // ---- FFN ----
    k_rmsnorm<<<ROWS, 256, 0, stream>>>(X, ff_gamma + (size_t)l * DIMM, nullptr, XNb);
    wtrans(w1, DIMM, FFD);
    gemm(XNb, b1, nullptr, nullptr, TB, ROWS, FFD, DIMM, 1);       // GELU
    wtrans(w2, FFD, DIMM);
    gemm(TB, b2, X, X, nullptr, ROWS, DIMM, FFD, 0);               // + residual
  }

  k_rmsnorm<<<ROWS, 256, 0, stream>>>(X, final_gamma, (float*)d_out, nullptr);
}
